// SRUppTransducerAttention_24670292149103
// MI455X (gfx1250) — compile-verified, matrix operand order checked by reading, not run on hardware
//
#include <hip/hip_runtime.h>
#include <stdint.h>
#include <stddef.h>

// ---------------------------------------------------------------------------
// SRUpp transducer attention block on MI455X (gfx1250, wave32, WMMA).
// bf16 staging: convert tensors to bf16 ONCE, then all matmuls are pure
// v_wmma_f32_16x16x32_bf16 fed by global_load_b128 / ds_load_b128.
// ---------------------------------------------------------------------------

#define DEV __device__ __forceinline__

typedef __attribute__((ext_vector_type(16))) __bf16 v16bf;
typedef __attribute__((ext_vector_type(8)))  float  v8f;

union BF16x16 { v16bf v; uint16_t u[16]; uint4 q[2]; };

// Round-to-nearest-even f32 -> bf16 (bit pattern).
DEV uint16_t f32_to_bf16_bits(float f) {
  union { float f; uint32_t u; } a; a.f = f;
  uint32_t r = a.u + 0x7FFFu + ((a.u >> 16) & 1u);
  return (uint16_t)(r >> 16);
}

DEV v8f wmma_bf16(v16bf a, v16bf b, v8f c) {
  // 8 args: (neg_a, A, neg_b, B, c_mod, C, reuse_a, reuse_b)
  return __builtin_amdgcn_wmma_f32_16x16x32_bf16(false, a, false, b,
                                                 (short)0, c, false, false);
}

// 16x32 16-bit A/B fragment from one bf16 row (ISA 7.12.2 layout):
//   lanes 0-15 : halves 0..7 = K k0+0..7,  halves 8..15 = K k0+16..23
//   lanes 16-31: halves 0..7 = K k0+8..15, halves 8..15 = K k0+24..31
// Two 16-byte contiguous chunks -> two b128 loads per fragment.
DEV v16bf load_frag_bf16(const uint16_t* __restrict__ row, int k0, int hi) {
  BF16x16 t;
  const uint16_t* p = row + k0 + 8 * hi;
  t.q[0] = *(const uint4*)p;
  t.q[1] = *(const uint4*)(p + 16);
  return t.v;
}

// ---------------------------------------------------------------------------
// Elementwise f32 -> bf16 (one-time staging). 8 elements / thread.
// ---------------------------------------------------------------------------
__global__ void __launch_bounds__(256)
cvt_f32_bf16(const float* __restrict__ src, uint16_t* __restrict__ dst, int n) {
  const int i = (blockIdx.x * 256 + threadIdx.x) * 8;
  if (i >= n) return;
  const float4 a = *(const float4*)(src + i);
  const float4 b = *(const float4*)(src + i + 4);
  union { uint16_t u[8]; uint4 q; } t;
  t.u[0] = f32_to_bf16_bits(a.x); t.u[1] = f32_to_bf16_bits(a.y);
  t.u[2] = f32_to_bf16_bits(a.z); t.u[3] = f32_to_bf16_bits(a.w);
  t.u[4] = f32_to_bf16_bits(b.x); t.u[5] = f32_to_bf16_bits(b.y);
  t.u[6] = f32_to_bf16_bits(b.z); t.u[7] = f32_to_bf16_bits(b.w);
  *(uint4*)(dst + i) = t.q;
}

// ---------------------------------------------------------------------------
// Register-blocked GEMM: C[M,N] = A[M,K] * W[N,K]^T (bf16 in, f32 accumulate).
// One wave -> 32x64 C macro-tile: 8 WMMAs per K-step from 6 fragments.
// Epilogues:
//   EPI=0 : Cf (f32)                                   [output projection]
//   EPI=1 : Cf (f32) + Cb (bf16 copy)                  [q: residual + reuse]
//   EPI=2 : n<512 -> Cb = K rows (bf16, [M][512])
//           n>=512-> Cb2 = V transposed (bf16, [b][h][hd][L])   [kv]
// ---------------------------------------------------------------------------
template <int EPI>
__global__ void __launch_bounds__(256)
gemm_bf16_rb(const uint16_t* __restrict__ A, const uint16_t* __restrict__ W,
             float* __restrict__ Cf, uint16_t* __restrict__ Cb,
             uint16_t* __restrict__ Cb2, int M, int N, int K) {
  const int lane = threadIdx.x & 31;
  const int wave = threadIdx.x >> 5;
  const int colT = N >> 6;
  const int tile = blockIdx.x * 8 + wave;
  if (tile >= (M >> 5) * colT) return;          // wave-uniform
  const int m0 = (tile / colT) * 32;
  const int n0 = (tile % colT) * 64;
  const int hi = lane >> 4, l15 = lane & 15;

  const uint16_t* ar0 = A + (size_t)(m0 + l15) * K;
  const uint16_t* ar1 = A + (size_t)(m0 + 16 + l15) * K;
  const uint16_t* wr0 = W + (size_t)(n0 + l15) * K;
  const uint16_t* wr1 = W + (size_t)(n0 + 16 + l15) * K;
  const uint16_t* wr2 = W + (size_t)(n0 + 32 + l15) * K;
  const uint16_t* wr3 = W + (size_t)(n0 + 48 + l15) * K;

  v8f acc[2][4] = {};
  for (int k0 = 0; k0 < K; k0 += 32) {
    __builtin_prefetch(ar0 + k0 + 256, 0, 0);   // global_prefetch next panel
    const v16bf a0 = load_frag_bf16(ar0, k0, hi);
    const v16bf a1 = load_frag_bf16(ar1, k0, hi);
    const v16bf b0 = load_frag_bf16(wr0, k0, hi);
    const v16bf b1 = load_frag_bf16(wr1, k0, hi);
    const v16bf b2 = load_frag_bf16(wr2, k0, hi);
    const v16bf b3 = load_frag_bf16(wr3, k0, hi);
    acc[0][0] = wmma_bf16(a0, b0, acc[0][0]);
    acc[0][1] = wmma_bf16(a0, b1, acc[0][1]);
    acc[0][2] = wmma_bf16(a0, b2, acc[0][2]);
    acc[0][3] = wmma_bf16(a0, b3, acc[0][3]);
    acc[1][0] = wmma_bf16(a1, b0, acc[1][0]);
    acc[1][1] = wmma_bf16(a1, b1, acc[1][1]);
    acc[1][2] = wmma_bf16(a1, b2, acc[1][2]);
    acc[1][3] = wmma_bf16(a1, b3, acc[1][3]);
  }

#pragma unroll
  for (int i = 0; i < 2; ++i)
#pragma unroll
    for (int j = 0; j < 4; ++j)
#pragma unroll
      for (int r = 0; r < 8; ++r) {
        const int m = m0 + 16 * i + 8 * hi + r;
        const int n = n0 + 16 * j + l15;
        const float v = acc[i][j][r];
        if (EPI == 0) {
          Cf[(size_t)m * N + n] = v;
        } else if (EPI == 1) {
          Cf[(size_t)m * N + n] = v;
          Cb[(size_t)m * N + n] = f32_to_bf16_bits(v);
        } else {                                 // EPI == 2: kv split
          if (n0 < 512) {
            Cb[(size_t)m * 512 + n] = f32_to_bf16_bits(v);
          } else {
            const int l = m >> 2, bb = m & 3;    // m = l*B + b, B = 4
            const int c = n - 512, hh = c >> 6, hd = c & 63;
            Cb2[((size_t)((bb * 8 + hh) * 64 + hd) << 11) + l] =
                f32_to_bf16_bits(v);
          }
        }
      }
}

// ---------------------------------------------------------------------------
// One 32-key attention block (two 16x16 S tiles). MASK=true only for the
// diagonal tail. Row sums of P are produced by an extra WMMA against an
// all-ones B fragment (lands directly in the per-row C layout) -- no
// cross-lane sum shuffles. P goes C-layout -> A-fragment via a row-major
// LDS tile: 16x u16 stores, 2x ds_load_b128 back.
// ---------------------------------------------------------------------------
template <bool MASK>
DEV void attn_block(int s0, int q0, int b, int h, int hi, int l15,
                    const uint16_t* __restrict__ Kb,
                    const uint16_t* __restrict__ vTh,
                    const v16bf& qf0, const v16bf& qf1, const v16bf& onesv,
                    v8f (&cacc)[4], v8f& sumacc, float (&rowmax)[8],
                    uint16_t* ldsPw) {
  constexpr int Lc = 2048, Bc = 4, DPROJ = 512, HDc = 64;
  constexpr float SCALING = 0.125f;              // HEAD_DIM^-0.5

  // ---- S = Q * K^T : two 16x16 tiles over 32 keys, K-depth 64 ----
  const uint16_t* krow0 = Kb + (size_t)((s0 + l15) * Bc + b) * DPROJ + h * HDc;
  const uint16_t* krow1 =
      Kb + (size_t)((s0 + 16 + l15) * Bc + b) * DPROJ + h * HDc;
  v8f S0 = {}, S1 = {};
  S0 = wmma_bf16(qf0, load_frag_bf16(krow0, 0, hi), S0);
  S0 = wmma_bf16(qf1, load_frag_bf16(krow0, 32, hi), S0);
  S1 = wmma_bf16(qf0, load_frag_bf16(krow1, 0, hi), S1);
  S1 = wmma_bf16(qf1, load_frag_bf16(krow1, 32, hi), S1);

  // ---- online softmax (rows live in 16-lane halves) ----
#pragma unroll
  for (int r = 0; r < 8; ++r) {
    float sc0 = S0[r] * SCALING;
    float sc1 = S1[r] * SCALING;
    if (MASK) {
      const int l = q0 + r + 8 * hi;
      if (s0 + l15 > l)      sc0 = -3.0e38f;
      if (s0 + 16 + l15 > l) sc1 = -3.0e38f;
    }
    float t = fmaxf(sc0, sc1);
    t = fmaxf(t, __shfl_xor(t, 1, 32));
    t = fmaxf(t, __shfl_xor(t, 2, 32));
    t = fmaxf(t, __shfl_xor(t, 4, 32));
    t = fmaxf(t, __shfl_xor(t, 8, 32));
    const float newmax = fmaxf(rowmax[r], t);
    const float corr = __expf(rowmax[r] - newmax);
    rowmax[r] = newmax;
    const float p0 = __expf(sc0 - newmax);
    const float p1 = __expf(sc1 - newmax);
    sumacc[r] *= corr;
#pragma unroll
    for (int nt = 0; nt < 4; ++nt) cacc[nt][r] *= corr;
    const int row = r + 8 * hi;
    ldsPw[row * 32 + l15]      = f32_to_bf16_bits(p0);
    ldsPw[row * 32 + 16 + l15] = f32_to_bf16_bits(p1);
  }
  asm volatile("s_wait_dscnt 0x0" ::: "memory");

  // ---- reload P as 16x32 A-fragment: two ds_load_b128 ----
  BF16x16 pf;
  pf.q[0] = *(const uint4*)&ldsPw[l15 * 32 + 8 * hi];
  pf.q[1] = *(const uint4*)&ldsPw[l15 * 32 + 16 + 8 * hi];

  // ---- row sums of P via the matrix pipe (B = ones) ----
  sumacc = wmma_bf16(pf.v, onesv, sumacc);

  // ---- ctx += P * V : V^T rows are contiguous along keys ----
#pragma unroll
  for (int nt = 0; nt < 4; ++nt) {
    const uint16_t* vrow = vTh + (size_t)(nt * 16 + l15) * Lc;
    const v16bf vf = load_frag_bf16(vrow, s0, hi);
    cacc[nt] = wmma_bf16(pf.v, vf, cacc[nt]);
  }
}

// ---------------------------------------------------------------------------
// Flash attention: 32 heads (B*H), head_dim 64, causal, online softmax.
// One wave per (head, 16-query tile). Unmasked fast path for all key blocks
// strictly below the diagonal; masked instantiation only for the tail.
// Epilogue fuses  out = ctx*alpha + q  and writes bf16 for the final GEMM.
// ---------------------------------------------------------------------------
__global__ void __launch_bounds__(256)
flash_attn(const uint16_t* __restrict__ Qb, const uint16_t* __restrict__ Kb,
           const uint16_t* __restrict__ VTb, const float* __restrict__ Qf32,
           const float* __restrict__ alpha_p, uint16_t* __restrict__ OutBf) {
  constexpr int Lc = 2048, Bc = 4, Hc = 8, HDc = 64, DPROJ = 512;
  __shared__ __align__(16) uint16_t ldsP[8][16 * 32];  // per-wave P tile

  const int lane = threadIdx.x & 31;
  const int wave = threadIdx.x >> 5;
  const int wid  = blockIdx.x * 8 + wave;
  const int qTiles = Lc / 16;                    // 128
  if (wid >= Bc * Hc * qTiles) return;
  const int head = wid / qTiles;
  const int q0   = (wid % qTiles) * 16;
  const int b = head / Hc, h = head % Hc;
  const int hi = lane >> 4, l15 = lane & 15;
  const float alpha = alpha_p[0];
  uint16_t* ldsPw = &ldsP[wave][0];

  // Q fragments (16 queries x 64 dims), unscaled; scale applied to scores.
  const uint16_t* qrow =
      Qb + (size_t)((q0 + l15) * Bc + b) * DPROJ + h * HDc;
  const v16bf qf0 = load_frag_bf16(qrow, 0, hi);
  const v16bf qf1 = load_frag_bf16(qrow, 32, hi);

  // all-ones B fragment for the row-sum WMMA
  BF16x16 ones;
#pragma unroll
  for (int i = 0; i < 16; ++i) ones.u[i] = 0x3F80;   // bf16(1.0)

  const uint16_t* vTh = VTb + (size_t)((b * Hc + h) * HDc) * Lc;

  v8f cacc[4] = {};
  v8f sumacc = {};
  float rowmax[8];
#pragma unroll
  for (int r = 0; r < 8; ++r) rowmax[r] = -3.0e38f;

  const int diagkb = q0 >> 5;                    // blocks fully below diagonal
  const int nkb = (q0 + 16 + 31) >> 5;           // total causal key blocks
  for (int kb = 0; kb < diagkb; ++kb)
    attn_block<false>(kb * 32, q0, b, h, hi, l15, Kb, vTh, qf0, qf1, ones.v,
                      cacc, sumacc, rowmax, ldsPw);
  for (int kb = diagkb; kb < nkb; ++kb)
    attn_block<true>(kb * 32, q0, b, h, hi, l15, Kb, vTh, qf0, qf1, ones.v,
                     cacc, sumacc, rowmax, ldsPw);

  // ---- normalize, fuse residual (out = ctx*alpha + q), store bf16 ----
#pragma unroll
  for (int r = 0; r < 8; ++r) {
    const int l = q0 + r + 8 * hi;
    const float inv = alpha / sumacc[r];
    const size_t idx = (size_t)(l * Bc + b) * DPROJ + h * HDc;
#pragma unroll
    for (int nt = 0; nt < 4; ++nt) {
      const int c = nt * 16 + l15;
      OutBf[idx + c] = f32_to_bf16_bits(cacc[nt][r] * inv + Qf32[idx + c]);
    }
  }
}

// ---------------------------------------------------------------------------
// Launch pipeline. Workspace (bytes, all 16B-aligned):
//   in_bf 16M | w1_bf 1M | w2_bf 1M | w3_bf 1M | q_f32 16M |
//   q_bf 8M | k_bf 8M | vT_bf 8M | out_bf 8M   (total 67 MB, L2-resident)
// ---------------------------------------------------------------------------
extern "C" void kernel_launch(void* const* d_in, const int* in_sizes, int n_in,
                              void* d_out, int out_size, void* d_ws, size_t ws_size,
                              hipStream_t stream) {
  (void)in_sizes; (void)n_in; (void)out_size; (void)ws_size;
  const float* input = (const float*)d_in[0];
  /* d_in[1] = attn_mask: strictly causal, computed analytically */
  const float* w1    = (const float*)d_in[2];
  const float* w2    = (const float*)d_in[3];
  const float* w3    = (const float*)d_in[4];
  const float* alpha = (const float*)d_in[5];
  float* out = (float*)d_out;

  constexpr int M = 2048 * 4;                  // L*B = 8192
  constexpr size_t SZ_IN = (size_t)M * 1024;   // input elements
  constexpr size_t SZ_W  = (size_t)512 * 1024; // each weight: 0.5M elements

  char* p = (char*)d_ws;
  uint16_t* in_bf = (uint16_t*)p;  p += SZ_IN * 2;
  uint16_t* w1_bf = (uint16_t*)p;  p += SZ_W * 2;
  uint16_t* w2_bf = (uint16_t*)p;  p += SZ_W * 2;
  uint16_t* w3_bf = (uint16_t*)p;  p += SZ_W * 2;
  float*    q_f32 = (float*)p;     p += (size_t)M * 512 * 4;
  uint16_t* q_bf  = (uint16_t*)p;  p += (size_t)M * 512 * 2;
  uint16_t* k_bf  = (uint16_t*)p;  p += (size_t)M * 512 * 2;
  uint16_t* vT_bf = (uint16_t*)p;  p += (size_t)M * 512 * 2;
  uint16_t* out_bf = (uint16_t*)p;

  // ---- one-time bf16 staging ----
  cvt_f32_bf16<<<SZ_IN / 2048, 256, 0, stream>>>(input, in_bf, (int)SZ_IN);
  cvt_f32_bf16<<<SZ_W / 2048, 256, 0, stream>>>(w1, w1_bf, (int)SZ_W);
  cvt_f32_bf16<<<SZ_W / 2048, 256, 0, stream>>>(w2, w2_bf, (int)SZ_W);
  cvt_f32_bf16<<<SZ_W / 2048, 256, 0, stream>>>(w3, w3_bf, (int)SZ_W);

  // ---- q = input @ w1^T (8192x512x1024), emit f32 + bf16 ----
  gemm_bf16_rb<1><<<(M / 32) * (512 / 64) / 8, 256, 0, stream>>>(
      in_bf, w1_bf, q_f32, q_bf, nullptr, M, 512, 1024);
  // ---- kv = q @ w2^T (8192x1024x512), emit K rows + V^T ----
  gemm_bf16_rb<2><<<(M / 32) * (1024 / 64) / 8, 256, 0, stream>>>(
      q_bf, w2_bf, nullptr, k_bf, vT_bf, M, 1024, 512);
  // ---- causal MHA + fused residual: out_bf = ctx*alpha + q ----
  flash_attn<<<(32 * 128) / 8, 256, 0, stream>>>(
      q_bf, k_bf, vT_bf, q_f32, alpha, out_bf);
  // ---- final = out @ w3^T (8192x1024x512) -> d_out f32 ----
  gemm_bf16_rb<0><<<(M / 32) * (1024 / 64) / 8, 256, 0, stream>>>(
      out_bf, w3_bf, out, nullptr, nullptr, M, 1024, 512);
}
